// EncodeProcessDecode_51333449122058
// MI455X (gfx1250) — compile-verified, matrix operand order checked by reading, not run on hardware
//
#include <hip/hip_runtime.h>
#include <hip/hip_bf16.h>

typedef __attribute__((ext_vector_type(16))) __bf16 v16bf;
typedef __attribute__((ext_vector_type(8)))  float  v8f;

#define HSTRIDE 132   // padded LDS row stride (floats) to avoid bank conflicts
#define MLP_WAVES 4

// ---------------------------------------------------------------------------
// Feature builders
// ---------------------------------------------------------------------------
__global__ void build_node_feat(const int* __restrict__ node_type,
                                const float* __restrict__ heat,
                                const float* __restrict__ temp,
                                const float* __restrict__ nmean,
                                const float* __restrict__ nstd,
                                float* __restrict__ nodeF, int N) {
  int i = blockIdx.x * blockDim.x + threadIdx.x;
  if (i >= N) return;
  float f[5];
  f[0] = heat[i];
  int t = node_type[i];
  f[1] = (t == 0) ? 1.f : 0.f;
  f[2] = (t == 1) ? 1.f : 0.f;
  f[3] = (t == 2) ? 1.f : 0.f;
  f[4] = temp[i];
  float* o = nodeF + (size_t)i * 32;
#pragma unroll
  for (int j = 0; j < 5; j++) o[j] = (f[j] - nmean[j]) / nstd[j];
#pragma unroll
  for (int j = 5; j < 32; j++) o[j] = 0.f;
}

__global__ void build_edge_feat(const float* __restrict__ mesh,
                                const float* __restrict__ temp,
                                const int* __restrict__ snd,
                                const int* __restrict__ rcv,
                                const float* __restrict__ emean,
                                const float* __restrict__ estd,
                                float* __restrict__ edgeF, int E) {
  int i = blockIdx.x * blockDim.x + threadIdx.x;
  if (i >= E) return;
  int s = snd[i], r = rcv[i];
  float rx = mesh[s * 3 + 0] - mesh[r * 3 + 0];
  float ry = mesh[s * 3 + 1] - mesh[r * 3 + 1];
  float rz = mesh[s * 3 + 2] - mesh[r * 3 + 2];
  float nn = sqrtf(rx * rx + ry * ry + rz * rz);
  float dT = temp[s] - temp[r];
  float f[5] = {rx, ry, rz, nn, dT};
  float* o = edgeF + (size_t)i * 32;
#pragma unroll
  for (int j = 0; j < 5; j++) o[j] = (f[j] - emean[j]) / estd[j];
#pragma unroll
  for (int j = 5; j < 32; j++) o[j] = 0.f;
}

__global__ void zero_f32(float* __restrict__ p, int n) {
  int i = blockIdx.x * blockDim.x + threadIdx.x;
  if (i < n) p[i] = 0.f;
}

// ---------------------------------------------------------------------------
// Weight pre-fragmentation: row-major f32 W[K][128] -> bf16 WMMA-B fragment
// order: frag[(c*8+nt)*32 + lane][e], e=0..15, matching the 16-bit operand
// layout of v_wmma_f32_16x16x32_bf16 (ISA 7.12.2). Zero-pads K to Kpad.
// ---------------------------------------------------------------------------
__global__ void pack_wfrag(const float* __restrict__ W, __bf16* __restrict__ wf,
                           int Ksrc, int Kpad) {
  int t = blockIdx.x * blockDim.x + threadIdx.x;
  int total = (Kpad >> 5) * 8 * 32;
  if (t >= total) return;
  int lane = t & 31;
  int nt = (t >> 5) & 7;
  int c = t >> 8;
  int m = lane & 15, hf = lane >> 4;
  int n = nt * 16 + m;
  __bf16* o = wf + (size_t)t * 16;
#pragma unroll
  for (int e = 0; e < 16; e++) {
    int k = c * 32 + ((e < 8) ? 0 : 16) + hf * 8 + (e & 7);
    float v = (k < Ksrc) ? W[(size_t)k * 128 + n] : 0.f;
    o[e] = (__bf16)v;
  }
}

// ---------------------------------------------------------------------------
// Generic fused MLP: X(rows x K) -> relu(X W1 + b1) -> relu(. W2 + b2) ->
// LayerNorm(g, be) -> optional residual add and atomic scatter-add.
// X is the concat of up to 3 source matrices (each 32*CPS floats wide), with
// optional row-index gather per source. One wave32 per 16-row tile; all
// matmul work through v_wmma_f32_16x16x32_bf16. KCH/CPS are compile-time so
// the layer-1 loop fully unrolls and B fragments are preloaded in batches of
// 8 to keep WMMAs from stalling on individual load round-trips.
// ---------------------------------------------------------------------------
template <int KCH, int CPS>
__global__ void __launch_bounds__(128) mlp_wmma_kernel(
    const float* __restrict__ src0, const float* __restrict__ src1,
    const float* __restrict__ src2, const int* __restrict__ idx0,
    const int* __restrict__ idx1, const int* __restrict__ idx2, int rows,
    const __bf16* __restrict__ w1f, const __bf16* __restrict__ w2f,
    const float* __restrict__ b1, const float* __restrict__ b2,
    const float* __restrict__ g, const float* __restrict__ be,
    const float* __restrict__ resid, float* __restrict__ out,
    const int* __restrict__ recv, float* __restrict__ aggr) {
  __shared__ float hbuf[MLP_WAVES][16 * HSTRIDE];
  const int lane = threadIdx.x & 31;
  const int wave = threadIdx.x >> 5;
  const int hf = lane >> 4;
  const int m = lane & 15;
  const int srcw = CPS * 32;
  const int ntile_count = (rows + 15) >> 4;
  const int tile = blockIdx.x * MLP_WAVES + wave;
  if (tile >= ntile_count) return;
  float* hb = hbuf[wave];
  const int base = tile * 16;
  const int row = base + m;
  const bool rv = row < rows;

  v8f zero = {};
  v8f acc[8];
#pragma unroll
  for (int nt = 0; nt < 8; nt++) acc[nt] = zero;

  // ---- layer 1 ----
  const v16bf* w1v = (const v16bf*)w1f;
#pragma unroll
  for (int c = 0; c < KCH; c++) {
    constexpr int dummy = 0;
    (void)dummy;
    const int s = c / CPS;           // compile-time folded
    const int kl = (c % CPS) * 32;   // compile-time folded
    const float* sp = (s == 0) ? src0 : ((s == 1) ? src1 : src2);
    const int* ip = (s == 0) ? idx0 : ((s == 1) ? idx1 : idx2);
    v16bf a;
    if (rv) {
      int r = ip ? ip[row] : row;
      const float* p = sp + (size_t)r * srcw + kl + hf * 8;
#pragma unroll
      for (int j = 0; j < 8; j++) {
        a[j] = (__bf16)p[j];
        a[8 + j] = (__bf16)p[16 + j];
      }
    } else {
#pragma unroll
      for (int j = 0; j < 16; j++) a[j] = (__bf16)0.f;
    }
    // Preload all 8 B fragments of this K-chunk, then issue the 8 WMMAs.
    v16bf bfr[8];
#pragma unroll
    for (int nt = 0; nt < 8; nt++) bfr[nt] = w1v[((c << 3) + nt) * 32 + lane];
#pragma unroll
    for (int nt = 0; nt < 8; nt++) {
      acc[nt] = __builtin_amdgcn_wmma_f32_16x16x32_bf16(
          false, a, false, bfr[nt], (short)0, acc[nt], false, false);
    }
  }

  // bias + relu, stash h to LDS (row-major, padded stride) for re-fragmenting
#pragma unroll
  for (int nt = 0; nt < 8; nt++) {
    float bb = b1[nt * 16 + m];
#pragma unroll
    for (int v = 0; v < 8; v++) {
      float x = fmaxf(acc[nt][v] + bb, 0.f);
      hb[(v + 8 * hf) * HSTRIDE + nt * 16 + m] = x;
    }
  }

  // ---- layer 2 (K = 128) ----
  v8f acc2[8];
#pragma unroll
  for (int nt = 0; nt < 8; nt++) acc2[nt] = zero;
  const v16bf* w2v = (const v16bf*)w2f;
#pragma unroll
  for (int c = 0; c < 4; c++) {
    v16bf a;
    const float* p = hb + m * HSTRIDE + c * 32 + hf * 8;
#pragma unroll
    for (int j = 0; j < 8; j++) {
      a[j] = (__bf16)p[j];
      a[8 + j] = (__bf16)p[16 + j];
    }
    v16bf bfr[8];
#pragma unroll
    for (int nt = 0; nt < 8; nt++) bfr[nt] = w2v[((c << 3) + nt) * 32 + lane];
#pragma unroll
    for (int nt = 0; nt < 8; nt++) {
      acc2[nt] = __builtin_amdgcn_wmma_f32_16x16x32_bf16(
          false, a, false, bfr[nt], (short)0, acc2[nt], false, false);
    }
  }

  // bias + relu
#pragma unroll
  for (int nt = 0; nt < 8; nt++) {
    float bb = b2[nt * 16 + m];
#pragma unroll
    for (int v = 0; v < 8; v++) acc2[nt][v] = fmaxf(acc2[nt][v] + bb, 0.f);
  }

  // ---- LayerNorm per output row (128 values spread over 16 lanes x 8 nt) ----
  float sum[8], sq[8];
#pragma unroll
  for (int v = 0; v < 8; v++) {
    float s = 0.f, q = 0.f;
#pragma unroll
    for (int nt = 0; nt < 8; nt++) {
      float x = acc2[nt][v];
      s += x;
      q += x * x;
    }
    sum[v] = s;
    sq[v] = q;
  }
#pragma unroll
  for (int off = 1; off < 16; off <<= 1) {
#pragma unroll
    for (int v = 0; v < 8; v++) {
      sum[v] += __shfl_xor(sum[v], off, 32);
      sq[v] += __shfl_xor(sq[v], off, 32);
    }
  }
  float mu[8], rstd[8];
#pragma unroll
  for (int v = 0; v < 8; v++) {
    float mm = sum[v] * (1.f / 128.f);
    float var = sq[v] * (1.f / 128.f) - mm * mm;
    mu[v] = mm;
    rstd[v] = rsqrtf(var + 1e-5f);
  }
#pragma unroll
  for (int nt = 0; nt < 8; nt++) {
    float gn = g[nt * 16 + m], bn = be[nt * 16 + m];
#pragma unroll
    for (int v = 0; v < 8; v++) {
      float y = (acc2[nt][v] - mu[v]) * rstd[v] * gn + bn;
      hb[(v + 8 * hf) * HSTRIDE + nt * 16 + m] = y;
    }
  }

  // ---- coalesced output: residual add + optional atomic scatter-add ----
  const int cb = lane * 4;
  for (int r = 0; r < 16; r++) {
    int row2 = base + r;
    if (row2 >= rows) break;
    const float* hy = hb + r * HSTRIDE + cb;
    float y0 = hy[0], y1 = hy[1], y2 = hy[2], y3 = hy[3];
    if (aggr) {
      int rr = recv[row2];
      float* ap = aggr + (size_t)rr * 128 + cb;
      atomicAdd(ap + 0, y0);
      atomicAdd(ap + 1, y1);
      atomicAdd(ap + 2, y2);
      atomicAdd(ap + 3, y3);
    }
    float* op = out + (size_t)row2 * 128 + cb;
    if (resid) {
      const float* rp = resid + (size_t)row2 * 128 + cb;
      y0 += rp[0];
      y1 += rp[1];
      y2 += rp[2];
      y3 += rp[3];
    }
    op[0] = y0;
    op[1] = y1;
    op[2] = y2;
    op[3] = y3;
  }
}

// ---------------------------------------------------------------------------
// Decoder: (128->8) swish (8->10), scaled by cumsum(dt) = t+1
// ---------------------------------------------------------------------------
__global__ void decoder_kernel(const float* __restrict__ nodes,
                               const float* __restrict__ W1,
                               const float* __restrict__ b1,
                               const float* __restrict__ W2,
                               const float* __restrict__ b2,
                               float* __restrict__ out, int N) {
  int i = blockIdx.x * blockDim.x + threadIdx.x;
  if (i >= N) return;
  const float* x = nodes + (size_t)i * 128;
  float h[8];
#pragma unroll
  for (int j = 0; j < 8; j++) h[j] = b1[j];
  for (int k = 0; k < 128; k++) {
    float xv = x[k];
#pragma unroll
    for (int j = 0; j < 8; j++) h[j] += xv * W1[k * 8 + j];
  }
#pragma unroll
  for (int j = 0; j < 8; j++) {
    float s = h[j];
    h[j] = s / (1.f + __expf(-s));  // swish
  }
#pragma unroll
  for (int t = 0; t < 10; t++) {
    float o = b2[t];
#pragma unroll
    for (int j = 0; j < 8; j++) o += h[j] * W2[j * 10 + t];
    out[(size_t)i * 10 + t] = o * (float)(t + 1);
  }
}

// ---------------------------------------------------------------------------
// Host launcher
// ---------------------------------------------------------------------------
extern "C" void kernel_launch(void* const* d_in, const int* in_sizes, int n_in,
                              void* d_out, int out_size, void* d_ws,
                              size_t ws_size, hipStream_t stream) {
  const int N = 15000, E = 120000, STEPS = 10;

  auto F = [&](int i) { return (const float*)d_in[i]; };
  auto I = [&](int i) { return (const int*)d_in[i]; };

  const int *node_type, *senders, *receivers;
  const float *heat, *temp, *mesh;
  const float *node_mean, *node_std, *edge_mean, *edge_std;
  const float *enW1, *enb1, *enW2, *enb2, *eng, *enbe;
  const float *eeW1, *eeb1, *eeW2, *eeb2, *eeg, *eebe;
  const float *beW1, *beb1, *beW2, *beb2, *beg, *bebe;
  const float *bnW1, *bnb1, *bnW2, *bnb2, *bng, *bnbe;
  const float *dW1, *db1, *dW2, *db2;

  // Disambiguate pytree flatten order by leaf sizes.
  if (n_in > 0 && in_sizes[0] == 1280) {
    // full alphabetical flatten
    bebe = F(0); beg = F(1); beW1 = F(2); beb1 = F(3); beW2 = F(4); beb2 = F(5);
    bnbe = F(6); bng = F(7); bnW1 = F(8); bnb1 = F(9); bnW2 = F(10); bnb2 = F(11);
    dW1 = F(12); dW2 = F(13); db1 = F(14); db2 = F(15);
    eebe = F(16); eeg = F(17); eeW1 = F(18); eeb1 = F(19); eeW2 = F(20); eeb2 = F(21);
    enbe = F(22); eng = F(23); enW1 = F(24); enb1 = F(25); enW2 = F(26); enb2 = F(27);
    heat = F(28); mesh = F(29); node_type = I(30);
    edge_mean = F(31); edge_std = F(32); node_mean = F(33); node_std = F(34);
    receivers = I(35); senders = I(36); temp = F(37);
  } else if (n_in > 10 && in_sizes[10] == 640) {
    // pure insertion order
    node_type = I(0); heat = F(1); temp = F(2); mesh = F(3);
    senders = I(4); receivers = I(5);
    node_mean = F(6); node_std = F(7); edge_mean = F(8); edge_std = F(9);
    enW1 = F(10); enb1 = F(11); enW2 = F(12); enb2 = F(13); eng = F(14); enbe = F(15);
    eeW1 = F(16); eeb1 = F(17); eeW2 = F(18); eeb2 = F(19); eeg = F(20); eebe = F(21);
    beW1 = F(22); beb1 = F(23); beW2 = F(24); beb2 = F(25); beg = F(26); bebe = F(27);
    bnW1 = F(28); bnb1 = F(29); bnW2 = F(30); bnb2 = F(31); bng = F(32); bnbe = F(33);
    dW1 = F(34); db1 = F(35); dW2 = F(36); db2 = F(37);
  } else {
    // positional args, jax-sorted within each pytree
    node_type = I(0); heat = F(1); temp = F(2); mesh = F(3);
    senders = I(4); receivers = I(5);
    edge_mean = F(6); edge_std = F(7); node_mean = F(8); node_std = F(9);
    eebe = F(10); eeg = F(11); eeW1 = F(12); eeb1 = F(13); eeW2 = F(14); eeb2 = F(15);
    enbe = F(16); eng = F(17); enW1 = F(18); enb1 = F(19); enW2 = F(20); enb2 = F(21);
    bebe = F(22); beg = F(23); beW1 = F(24); beb1 = F(25); beW2 = F(26); beb2 = F(27);
    bnbe = F(28); bng = F(29); bnW1 = F(30); bnb1 = F(31); bnW2 = F(32); bnb2 = F(33);
    dW1 = F(34); dW2 = F(35); db1 = F(36); db2 = F(37);
  }

  // ---- workspace carve ----
  char* ws = (char*)d_ws;
  size_t off = 0;
  auto carve = [&](size_t bytes) -> void* {
    void* p = ws + off;
    off = (off + bytes + 255) & ~(size_t)255;
    return p;
  };
  float* nodes = (float*)carve((size_t)N * 128 * 4);
  float* edges = (float*)carve((size_t)E * 128 * 4);
  float* aggr = (float*)carve((size_t)N * 128 * 4);
  float* nodeF = (float*)carve((size_t)N * 32 * 4);
  float* edgeF = (float*)carve((size_t)E * 32 * 4);
  __bf16* w1frag = (__bf16*)carve((size_t)384 * 128 * 2);
  __bf16* w2frag = (__bf16*)carve((size_t)128 * 128 * 2);
  if (off > ws_size) return;  // insufficient scratch

  const int tilesN = (N + 15) >> 4, tilesE = (E + 15) >> 4;
  const int blkN = (tilesN + MLP_WAVES - 1) / MLP_WAVES;
  const int blkE = (tilesE + MLP_WAVES - 1) / MLP_WAVES;
  auto packBlocks = [](int Kpad) { return ((Kpad >> 5) * 256 + 255) / 256; };

  // ---- encoder ----
  build_node_feat<<<(N + 255) / 256, 256, 0, stream>>>(
      node_type, heat, temp, node_mean, node_std, nodeF, N);
  build_edge_feat<<<(E + 255) / 256, 256, 0, stream>>>(
      mesh, temp, senders, receivers, edge_mean, edge_std, edgeF, E);

  pack_wfrag<<<packBlocks(32), 256, 0, stream>>>(enW1, w1frag, 5, 32);
  pack_wfrag<<<packBlocks(128), 256, 0, stream>>>(enW2, w2frag, 128, 128);
  mlp_wmma_kernel<1, 1><<<blkN, 128, 0, stream>>>(
      nodeF, nullptr, nullptr, nullptr, nullptr, nullptr, N, w1frag, w2frag,
      enb1, enb2, eng, enbe, nullptr, nodes, nullptr, nullptr);

  pack_wfrag<<<packBlocks(32), 256, 0, stream>>>(eeW1, w1frag, 5, 32);
  pack_wfrag<<<packBlocks(128), 256, 0, stream>>>(eeW2, w2frag, 128, 128);
  mlp_wmma_kernel<1, 1><<<blkE, 128, 0, stream>>>(
      edgeF, nullptr, nullptr, nullptr, nullptr, nullptr, E, w1frag, w2frag,
      eeb1, eeb2, eeg, eebe, nullptr, edges, nullptr, nullptr);

  // ---- processor: 10 message-passing steps ----
  for (int s = 0; s < STEPS; s++) {
    zero_f32<<<(N * 128 + 255) / 256, 256, 0, stream>>>(aggr, N * 128);

    // edge MLP: concat(n[snd], n[rcv], e) @ W1(384x128) ... + residual,
    // plus atomic segment-sum of new_e into aggr[receivers]
    pack_wfrag<<<packBlocks(384), 256, 0, stream>>>(
        beW1 + (size_t)s * 384 * 128, w1frag, 384, 384);
    pack_wfrag<<<packBlocks(128), 256, 0, stream>>>(
        beW2 + (size_t)s * 128 * 128, w2frag, 128, 128);
    mlp_wmma_kernel<12, 4><<<blkE, 128, 0, stream>>>(
        nodes, nodes, edges, senders, receivers, nullptr, E, w1frag, w2frag,
        beb1 + s * 128, beb2 + s * 128, beg + s * 128, bebe + s * 128, edges,
        edges, receivers, aggr);

    // node MLP: concat(n, aggr) @ W1(256x128) ... + residual
    pack_wfrag<<<packBlocks(256), 256, 0, stream>>>(
        bnW1 + (size_t)s * 256 * 128, w1frag, 256, 256);
    pack_wfrag<<<packBlocks(128), 256, 0, stream>>>(
        bnW2 + (size_t)s * 128 * 128, w2frag, 128, 128);
    mlp_wmma_kernel<8, 4><<<blkN, 128, 0, stream>>>(
        nodes, aggr, nullptr, nullptr, nullptr, nullptr, N, w1frag, w2frag,
        bnb1 + s * 128, bnb2 + s * 128, bng + s * 128, bnbe + s * 128, nodes,
        nodes, nullptr, nullptr);
  }

  // ---- decoder ----
  decoder_kernel<<<(N + 127) / 128, 128, 0, stream>>>(
      nodes, dW1, db1, dW2, db2, (float*)d_out, N);
}
